// MultiHeadAttention_12051678233305
// MI455X (gfx1250) — matrix operand order repaired and checked by reading, not compiled, not run on hardware
//
#include <hip/hip_runtime.h>
#include <hip/hip_bf16.h>
#include <stdint.h>

// ---------------- types ----------------
typedef __bf16 bf16_t;
typedef __attribute__((ext_vector_type(16))) __bf16 v16bf;
typedef __attribute__((ext_vector_type(8)))  float  v8f;

#define N_HEAD  16
#define D_MODEL 1024
#define BATCH   2
#define SEQ     2048
#define DK      64
#define M_TOTAL (BATCH*SEQ)      // 4096 rows

union Frag16 { v16bf v; uint4 u[2]; };

// lane pair (l <-> l^16) exchange via ds_swizzle SWAPX16 (xor=0x10,and=0x1f)
__device__ __forceinline__ float lane_xor16(float x) {
  int i = __builtin_amdgcn_ds_swizzle(__builtin_bit_cast(int, x), 0x401F);
  return __builtin_bit_cast(float, i);
}

// ---------------- prep kernels ----------------
__global__ void cvt_f32_to_bf16(const float* __restrict__ in,
                                bf16_t* __restrict__ out, int n) {
  int i = blockIdx.x * blockDim.x + threadIdx.x;
  if (i < n) out[i] = (bf16_t)in[i];
}

// W: [K=1024][N=1024] f32 row-major  ->  Wt: [N][K] bf16 (B^T for WMMA B-frags)
__global__ void transpose_to_bf16(const float* __restrict__ W,
                                  bf16_t* __restrict__ Wt) {
  int i = blockIdx.x * blockDim.x + threadIdx.x;   // 1M threads
  int n = i >> 10, k = i & 1023;                   // write contiguous in k
  Wt[n * 1024 + k] = (bf16_t)W[k * 1024 + n];
}

// ---------------- fragment loaders ----------------
__device__ __forceinline__ void load_a_frag(Frag16& f, const bf16_t* rowp, int kk, int khalf) {
  const bf16_t* p = rowp + kk + khalf;       // 16B aligned
  f.u[0] = *(const uint4*)(p);               // K = kk+khalf .. +7
  f.u[1] = *(const uint4*)(p + 8);           // K = kk+khalf+8 .. +15
}
__device__ __forceinline__ void load_b_frag(Frag16& f, const bf16_t* colp, int kk, int kg) {
  const bf16_t* p = colp + kk + kg;          // 16B aligned
  f.u[0] = *(const uint4*)(p);               // K = kk+kg .. +7
  f.u[1] = *(const uint4*)(p + 8);           // K = kk+kg+8 .. +15
}

// ---------------- WMMA GEMM:  Out[m][n] = A[m][:] . Bt[n][:] + bias[n] --------
// MODE 0: Q  -> bf16, scaled by 1/sqrt(dk)=0.125, row-major [4096][1024]
// MODE 1: K  -> bf16 row-major [4096][1024]
// MODE 2: V  -> bf16 transposed per head: [b][h][64][2048]
// MODE 3: O  -> f32 row-major [4096][1024] (final output)
template<int MODE>
__global__ __launch_bounds__(256) void gemm_wmma_bf16(
    const bf16_t* __restrict__ A,    // [M][1024]
    const bf16_t* __restrict__ Bt,   // [1024][1024]
    const float*  __restrict__ bias, // [1024]
    void* __restrict__ Out)
{
  const int lane = threadIdx.x & 31;
  const int wid  = blockIdx.x * (blockDim.x >> 5) + (threadIdx.x >> 5);
  const int mt   = wid >> 4;          // 0..255
  const int nt   = wid & 15;          // 0..15  (64-wide N tile)
  const int m0   = mt * 16;
  const int n0   = nt * 64;

  const int half  = lane >> 4;        // 0/1
  const int lidx  = lane & 15;
  const int khalf = half * 16;        // A-frag K sub-offset
  const int kg    = half * 16;        // B-frag K sub-offset

  v8f acc[4];
#pragma unroll
  for (int s = 0; s < 4; ++s) acc[s] = (v8f)0.f;

  const bf16_t* arow = A + (size_t)(m0 + lidx) * 1024;
  const bf16_t* bcol[4];
#pragma unroll
  for (int s = 0; s < 4; ++s) bcol[s] = Bt + (size_t)(n0 + s * 16 + lidx) * 1024;

  // two named fragment sets (NO runtime buffer index -> stays in VGPRs)
  Frag16 afA, afB, bA[4], bB[4];
  load_a_frag(afA, arow, 0, khalf);
#pragma unroll
  for (int s = 0; s < 4; ++s) load_b_frag(bA[s], bcol[s], 0, kg);

  // K-loop unrolled x2: 64 elements per iteration, 16 iterations
  for (int kk = 0; kk < 1024; kk += 64) {
    // issue B-set loads (kk+32) before computing on A-set
    load_a_frag(afB, arow, kk + 32, khalf);
#pragma unroll
    for (int s = 0; s < 4; ++s) load_b_frag(bB[s], bcol[s], kk + 32, kg);
#pragma unroll
    for (int s = 0; s < 4; ++s)
      acc[s] = __builtin_amdgcn_wmma_f32_16x16x32_bf16(
          false, afA.v, false, bA[s].v, (short)0, acc[s], false, false);
    // issue next A-set loads (kk+64) before computing on B-set
    if (kk + 64 < 1024) {
      load_a_frag(afA, arow, kk + 64, khalf);
#pragma unroll
      for (int s = 0; s < 4; ++s) load_b_frag(bA[s], bcol[s], kk + 64, kg);
    }
#pragma unroll
    for (int s = 0; s < 4; ++s)
      acc[s] = __builtin_amdgcn_wmma_f32_16x16x32_bf16(
          false, afB.v, false, bB[s].v, (short)0, acc[s], false, false);
  }

  const int rowbase = half * 8;
#pragma unroll
  for (int s = 0; s < 4; ++s) {
    const int col = n0 + s * 16 + lidx;
    const float bcst = bias[col];
#pragma unroll
    for (int r = 0; r < 8; ++r) {
      int m   = m0 + rowbase + r;
      float v = acc[s][r] + bcst;
      if (MODE == 0) {
        ((bf16_t*)Out)[(size_t)m * 1024 + col] = (bf16_t)(v * 0.125f);
      } else if (MODE == 1) {
        ((bf16_t*)Out)[(size_t)m * 1024 + col] = (bf16_t)v;
      } else if (MODE == 2) {
        int b = m >> 11, t = m & 2047;
        int h = col >> 6, d = col & 63;
        ((bf16_t*)Out)[(size_t)(((b * N_HEAD + h) * DK + d) << 11) + t] = (bf16_t)v;
      } else {
        ((float*)Out)[(size_t)m * 1024 + col] = v;
      }
    }
  }
}

// ---------------- flash attention helpers ----------------
__device__ __forceinline__ void load_k_chunk(Frag16 (&kf)[4], const bf16_t* Kbase,
                                             int kb, int lidx, int half) {
#pragma unroll
  for (int t = 0; t < 2; ++t)
#pragma unroll
    for (int s = 0; s < 2; ++s) {
      const bf16_t* p = Kbase + (size_t)(kb + t * 16 + lidx) * D_MODEL + s * 32 + half * 16;
      kf[t * 2 + s].u[0] = *(const uint4*)(p);
      kf[t * 2 + s].u[1] = *(const uint4*)(p + 8);
    }
}

// one 32-key chunk; kaCur/kaNxt are distinct named buffers (constant indexing only)
template<bool CAUSAL>
__device__ __forceinline__ void attn_chunk(
    int kb, int nk, int qglob, int half, int lidx, int lane,
    const bf16_t* __restrict__ Kbase, const bf16_t* __restrict__ Vbase,
    const unsigned char* __restrict__ vm,
    const Frag16 (&qb)[2], const Frag16 (&kaCur)[4], Frag16 (&kaNxt)[4],
    v8f (&acc)[4], float& mrow, float& lrow)
{
  // value-mask bits for the 32 keys of this chunk
  unsigned vmbits = (unsigned)__ballot(vm[kb + lane] != 0);

  // issue V fragment loads early: consumed only after softmax
  Frag16 vb[4];
#pragma unroll
  for (int dsub = 0; dsub < 4; ++dsub) {
    const bf16_t* vp = Vbase + (size_t)(dsub * 16 + lidx) * SEQ + kb + half * 8;
    vb[dsub].u[0] = *(const uint4*)(vp);
    vb[dsub].u[1] = *(const uint4*)(vp + 16);
  }
  // issue next chunk's K fragment loads (overlap with this chunk's compute)
  if (kb + 32 < nk) load_k_chunk(kaNxt, Kbase, kb + 32, lidx, half);

  // S^T tiles: c0 = keys kb..kb+15, c1 = keys kb+16..kb+31 (cols = queries)
  v8f c0 = (v8f)0.f, c1 = (v8f)0.f;
#pragma unroll
  for (int s = 0; s < 2; ++s) {
    c0 = __builtin_amdgcn_wmma_f32_16x16x32_bf16(
        false, kaCur[0 * 2 + s].v, false, qb[s].v, (short)0, c0, false, false);
    c1 = __builtin_amdgcn_wmma_f32_16x16x32_bf16(
        false, kaCur[1 * 2 + s].v, false, qb[s].v, (short)0, c1, false, false);
  }

  // mask + online softmax (rows = keys in regs, pair-lane reduce over 16<->16)
  float p0[8], p1[8];
  float tmax = -1e30f;
#pragma unroll
  for (int r = 0; r < 8; ++r) {
    int bit0 = half * 8 + r;           // key offset within chunk, tile 0
    float s0 = c0[r], s1 = c1[r];
    bool ok0 = (vmbits >> bit0) & 1u;
    bool ok1 = (vmbits >> (bit0 + 16)) & 1u;
    if (CAUSAL) {
      ok0 = ok0 && (kb + bit0 <= qglob);
      ok1 = ok1 && (kb + bit0 + 16 <= qglob);
    }
    s0 = ok0 ? s0 : s0 - 1e9f;         // reference: score - NEG*(~mask)
    s1 = ok1 ? s1 : s1 - 1e9f;
    p0[r] = s0; p1[r] = s1;
    tmax = fmaxf(tmax, fmaxf(s0, s1));
  }
  tmax = fmaxf(tmax, lane_xor16(tmax));
  float mnew = fmaxf(mrow, tmax);
  float corr = __expf(mrow - mnew);
  float tsum = 0.f;
#pragma unroll
  for (int r = 0; r < 8; ++r) {
    p0[r] = __expf(p0[r] - mnew);
    p1[r] = __expf(p1[r] - mnew);
    tsum += p0[r] + p1[r];
  }
  tsum += lane_xor16(tsum);
  lrow = lrow * corr + tsum;
  mrow = mnew;
#pragma unroll
  for (int s = 0; s < 4; ++s)
#pragma unroll
    for (int r = 0; r < 8; ++r) acc[s][r] *= corr;

  // S^T C-layout maps exactly onto P A-fragment layout: just pack to bf16.
  Frag16 pa;
#pragma unroll
  for (int r = 0; r < 8; ++r) {
    pa.v[r]     = (bf16_t)p0[r];   // keys kb + half*8 + r      (A K 0..7 / 8..15)
    pa.v[8 + r] = (bf16_t)p1[r];   // keys kb + 16 + half*8 + r (A K 16..23 / 24..31)
  }

  // ctx += P(16x32) x V(32x16) for 4 dim sub-tiles (V already in flight)
#pragma unroll
  for (int dsub = 0; dsub < 4; ++dsub)
    acc[dsub] = __builtin_amdgcn_wmma_f32_16x16x32_bf16(
        false, pa.v, false, vb[dsub].v, (short)0, acc[dsub], false, false);
}

// ---------------- flash attention: one wave per (b, h, 16-query tile) --------
__global__ __launch_bounds__(256) void flash_attn(
    const bf16_t* __restrict__ Q,      // [4096][1024], pre-scaled
    const bf16_t* __restrict__ Kmat,   // [4096][1024]
    const bf16_t* __restrict__ Vt,     // [b][h][64][2048]
    const unsigned char* __restrict__ qmask,  // [B][T]
    const unsigned char* __restrict__ vmask,  // [B][T]
    bf16_t* __restrict__ ctx)          // [4096][1024]
{
  const int lane = threadIdx.x & 31;
  const int wid  = blockIdx.x * (blockDim.x >> 5) + (threadIdx.x >> 5);
  const int qt = wid & 127;
  const int h  = (wid >> 7) & 15;
  const int b  = wid >> 11;
  const int q0 = qt * 16;

  const int half = lane >> 4;
  const int lidx = lane & 15;
  const int qglob = q0 + lidx;               // this lane's query (S^T column)

  // Q as B-fragments for the two dk k-steps (contiguous 32B per lane)
  Frag16 qb[2];
  {
    const bf16_t* qrow = Q + (size_t)(b * SEQ + q0 + lidx) * D_MODEL
                           + h * DK + half * 16;
    qb[0].u[0] = *(const uint4*)(qrow);
    qb[0].u[1] = *(const uint4*)(qrow + 8);
    qb[1].u[0] = *(const uint4*)(qrow + 32);
    qb[1].u[1] = *(const uint4*)(qrow + 40);
  }

  v8f acc[4];
#pragma unroll
  for (int s = 0; s < 4; ++s) acc[s] = (v8f)0.f;
  float mrow = -1e30f, lrow = 0.f;

  const int nk = q0 + 16;                              // causal key count
  const bf16_t* Kbase = Kmat + (size_t)(b * SEQ) * D_MODEL + h * DK;
  const bf16_t* Vbase = Vt + (size_t)((b * N_HEAD + h) * DK) * SEQ;
  const unsigned char* vm = vmask + b * SEQ;

  // two named K-fragment buffers, roles fixed per call site
  Frag16 ka0[4], ka1[4];
  load_k_chunk(ka0, Kbase, 0, lidx, half);

  // Chunk schedule: with 32-key chunks and 16-aligned query tiles there is
  // always exactly ONE causal chunk (the last). Main loop handles pairs of
  // non-causal chunks with fixed ping-pong roles; then an optional single
  // non-causal chunk; then the single causal chunk.
  int kb = 0;
  for (; kb + 63 <= q0; kb += 64) {
    attn_chunk<false>(kb,      nk, qglob, half, lidx, lane, Kbase, Vbase, vm,
                      qb, ka0, ka1, acc, mrow, lrow);
    attn_chunk<false>(kb + 32, nk, qglob, half, lidx, lane, Kbase, Vbase, vm,
                      qb, ka1, ka0, acc, mrow, lrow);
  }
  if (kb + 31 <= q0) {
    attn_chunk<false>(kb, nk, qglob, half, lidx, lane, Kbase, Vbase, vm,
                      qb, ka0, ka1, acc, mrow, lrow);
    kb += 32;
    // causal chunk reads ka1
    attn_chunk<true>(kb, nk, qglob, half, lidx, lane, Kbase, Vbase, vm,
                     qb, ka1, ka0, acc, mrow, lrow);
  } else {
    // causal chunk reads ka0
    attn_chunk<true>(kb, nk, qglob, half, lidx, lane, Kbase, Vbase, vm,
                     qb, ka0, ka1, acc, mrow, lrow);
  }

  // epilogue: 1/l and query mask live per-column; redistribute to C rows
  float invl = 1.0f / lrow;
  invl = (qmask[b * SEQ + qglob] != 0) ? invl : 0.f;
  float wq[8];
#pragma unroll
  for (int r = 0; r < 8; ++r) {
    int srcq = half * 8 + r;                           // lane srcq holds query srcq
    int ii = __builtin_amdgcn_ds_bpermute(srcq << 2, __builtin_bit_cast(int, invl));
    wq[r] = __builtin_bit_cast(float, ii);
  }
#pragma unroll
  for (int dsub = 0; dsub < 4; ++dsub) {
#pragma unroll
    for (int r = 0; r < 8; ++r) {
      int q   = q0 + half * 8 + r;
      int col = h * DK + dsub * 16 + lidx;             // head-major concat
      ctx[(size_t)(b * SEQ + q) * D_MODEL + col] = (bf16_t)(acc[dsub][r] * wq[r]);
    }
  }
}

// ---------------- host launcher ----------------
extern "C" void kernel_launch(void* const* d_in, const int* in_sizes, int n_in,
                              void* d_out, int out_size, void* d_ws, size_t ws_size,
                              hipStream_t stream) {
  const float* x  = (const float*)d_in[0];
  const unsigned char* qmask = (const unsigned char*)d_in[1];
  const unsigned char* vmask = (const unsigned char*)d_in[2];
  const float* Wq = (const float*)d_in[3];
  const float* bq = (const float*)d_in[4];
  const float* Wk = (const float*)d_in[5];
  const float* bk = (const float*)d_in[6];
  const float* Wv = (const float*)d_in[7];
  const float* bv = (const float*)d_in[8];
  const float* Wo = (const float*)d_in[9];
  const float* bo = (const float*)d_in[10];

  char* ws = (char*)d_ws;
  size_t off = 0;
  bf16_t* xb   = (bf16_t*)(ws + off); off += (size_t)M_TOTAL * D_MODEL * 2;
  bf16_t* Wqt  = (bf16_t*)(ws + off); off += (size_t)D_MODEL * D_MODEL * 2;
  bf16_t* Wkt  = (bf16_t*)(ws + off); off += (size_t)D_MODEL * D_MODEL * 2;
  bf16_t* Wvt  = (bf16_t*)(ws + off); off += (size_t)D_MODEL * D_MODEL * 2;
  bf16_t* Wot  = (bf16_t*)(ws + off); off += (size_t)D_MODEL * D_MODEL * 2;
  bf16_t* Qb   = (bf16_t*)(ws + off); off += (size_t)M_TOTAL * D_MODEL * 2;
  bf16_t* Kb   = (bf16_t*)(ws + off); off += (size_t)M_TOTAL * D_MODEL * 2;
  bf16_t* Vtb  = (bf16_t*)(ws + off); off += (size_t)M_TOTAL * D_MODEL * 2;
  bf16_t* ctxb = (bf16_t*)(ws + off); off += (size_t)M_TOTAL * D_MODEL * 2;

  // 1) precision prep
  {
    int n = M_TOTAL * D_MODEL;
    cvt_f32_to_bf16<<<(n + 255) / 256, 256, 0, stream>>>(x, xb, n);
  }
  {
    int blocks = (D_MODEL * D_MODEL) / 256;
    transpose_to_bf16<<<blocks, 256, 0, stream>>>(Wq, Wqt);
    transpose_to_bf16<<<blocks, 256, 0, stream>>>(Wk, Wkt);
    transpose_to_bf16<<<blocks, 256, 0, stream>>>(Wv, Wvt);
    transpose_to_bf16<<<blocks, 256, 0, stream>>>(Wo, Wot);
  }

  // 2) projections: 4096 waves (16x64 tiles) -> 512 blocks of 8 waves
  const int gemm_blocks = (M_TOTAL / 16) * (D_MODEL / 64) / 8;  // 512
  gemm_wmma_bf16<0><<<gemm_blocks, 256, 0, stream>>>(xb, Wqt, bq, (void*)Qb);
  gemm_wmma_bf16<1><<<gemm_blocks, 256, 0, stream>>>(xb, Wkt, bk, (void*)Kb);
  gemm_wmma_bf16<2><<<gemm_blocks, 256, 0, stream>>>(xb, Wvt, bv, (void*)Vtb);

  // 3) flash attention: B*H*(T/16) = 4096 waves -> 512 blocks
  flash_attn<<<512, 256, 0, stream>>>(Qb, Kb, Vtb, qmask, vmask, ctxb);

  // 4) output projection -> f32 d_out
  gemm_wmma_bf16<3><<<gemm_blocks, 256, 0, stream>>>(ctxb, Wot, bo, d_out);
}
